// NodeToNode_90400471646657
// MI455X (gfx1250) — matrix operand order repaired and checked by reading, not compiled, not run on hardware
//
#include <hip/hip_runtime.h>

#define NN      50000
#define DIN     128
#define DHID    256
#define DOUT    128
#define NEDGE   600000

typedef __attribute__((ext_vector_type(16))) _Float16 v16h;
typedef __attribute__((ext_vector_type(8)))  _Float16 v8h;
typedef __attribute__((ext_vector_type(8)))  float    v8f;

// ---------------------------------------------------------------- utilities
__device__ __forceinline__ float gelu_exact(float x) {
    // 0.5 * x * (1 + erf(x / sqrt(2)))
    return 0.5f * x * (1.0f + erff(x * 0.70710678118654752f));
}

// A-fragment (16x32 f16, MxK) from row-major LDS buffer.
// Lane L(0..15) = row L, K-chunks {kt*32+0..7, kt*32+16..23};
// Lane L+16     = row L, K-chunks {kt*32+8..15, kt*32+24..31}.
__device__ __forceinline__ v16h load_a_frag(const _Float16* buf, int stride,
                                            int kt, int hh, int l) {
    int c0 = kt * 32 + hh * 8;
    v8h lo = *(const v8h*)(buf + l * stride + c0);
    v8h hi = *(const v8h*)(buf + l * stride + c0 + 16);
    return __builtin_shufflevector(lo, hi, 0,1,2,3,4,5,6,7,8,9,10,11,12,13,14,15);
}

// B-fragment from pre-swizzled weights: fragment f = kt*NT+nt, 512 halves,
// lane chunk of 16 halves (32B) contiguous.
__device__ __forceinline__ v16h load_b_frag(const _Float16* pw, int NT,
                                            int kt, int nt, int lane) {
    return *(const v16h*)(pw + (((kt * NT + nt) << 9) + (lane << 4)));
}

// ---------------------------------------------------------------- kernels
__global__ void zero_f4_kernel(float4* __restrict__ p, int n4) {
    int i = blockIdx.x * blockDim.x + threadIdx.x;
    if (i < n4) p[i] = make_float4(0.f, 0.f, 0.f, 0.f);
}

// Pack K x N f32 weight into WMMA-B swizzled f16:
// dest half index = (kt*NT+nt)*512 + lane*16 + i,
// source = W[(kt*32 + (lane>>4)*8 + (i&7) + (i>>3)*16) * N + nt*16 + (lane&15)]
__global__ void pack_weight_kernel(const float* __restrict__ W,
                                   _Float16* __restrict__ pd, int K, int N) {
    int tid = blockIdx.x * blockDim.x + threadIdx.x;
    if (tid >= K * N) return;
    int NT = N >> 4;
    int f  = tid >> 9;
    int r  = tid & 511;
    int L  = r >> 4;
    int i  = r & 15;
    int kt = f / NT;
    int nt = f - kt * NT;
    int k  = kt * 32 + ((L >> 4) << 3) + (i & 7) + ((i >> 3) << 4);
    int n  = nt * 16 + (L & 15);
    pd[tid] = (_Float16)W[k * N + n];
}

// One 32-lane group per edge; lane owns 4 consecutive features.
// agg[r] += x[s]; agg[s] += x[r]   (bidirectional message passing)
__global__ void scatter_kernel(const float* __restrict__ x,
                               const int* __restrict__ ei,
                               float* __restrict__ agg, int nEdges) {
    long long gid = (long long)blockIdx.x * blockDim.x + threadIdx.x;
    int e = (int)(gid >> 5);
    int l = (int)(gid & 31);
    if (e >= nEdges) return;
    int s = ei[e];
    int r = ei[nEdges + e];
    const float4 xs = *(const float4*)(x + (long long)s * DIN + l * 4);
    const float4 xr = *(const float4*)(x + (long long)r * DIN + l * 4);
    float* ar = agg + (long long)r * DIN + l * 4;
    float* as = agg + (long long)s * DIN + l * 4;
    atomicAdd(ar + 0, xs.x); atomicAdd(ar + 1, xs.y);
    atomicAdd(ar + 2, xs.z); atomicAdd(ar + 3, xs.w);
    atomicAdd(as + 0, xr.x); atomicAdd(as + 1, xr.y);
    atomicAdd(as + 2, xr.z); atomicAdd(as + 3, xr.w);
}

// Fused 3-layer MLP (WMMA f16->f32) + exact GELU + LayerNorm.
// 4 waves/block, each wave owns 16 nodes end-to-end; LDS ping-pong 2x8KB/wave.
__global__ __launch_bounds__(128) void mlp_kernel(
        const float* __restrict__ agg,
        const _Float16* __restrict__ pw1, const _Float16* __restrict__ pw2,
        const _Float16* __restrict__ pw3,
        const float* __restrict__ b1, const float* __restrict__ b2,
        const float* __restrict__ b3,
        const float* __restrict__ lng, const float* __restrict__ lnb,
        float* __restrict__ out) {
    __shared__ _Float16 smem[4 * 2 * 16 * 256];   // 64 KB

    const int wave = threadIdx.x >> 5;
    const int lane = threadIdx.x & 31;
    const int hh   = lane >> 4;      // half-wave: 0 -> rows 0..7, 1 -> rows 8..15
    const int l    = lane & 15;      // column-within-tile / row-within-half

    _Float16* buf0 = smem + wave * (2 * 16 * 256);
    _Float16* buf1 = buf0 + 16 * 256;
    const int row0 = (blockIdx.x * 4 + wave) * 16;

    // -------- stage A: agg[row0..row0+15][0..127] f32 -> f16 (row-major) ----
    #pragma unroll
    for (int t = 0; t < 16; ++t) {
        int idx = t * 32 + lane;          // 512 float4-chunks total
        int r   = idx >> 5;               // 32 chunks per row
        int c4  = (idx & 31) << 2;
        float4 v = make_float4(0.f, 0.f, 0.f, 0.f);
        int gr = row0 + r;
        if (gr < NN) v = *(const float4*)(agg + (long long)gr * DIN + c4);
        _Float16* d = buf0 + r * DIN + c4;
        d[0] = (_Float16)v.x; d[1] = (_Float16)v.y;
        d[2] = (_Float16)v.z; d[3] = (_Float16)v.w;
    }

    // -------- layer 1: h1 = gelu(A(16x128) @ W1(128x256) + b1) -> buf1 ------
    #pragma unroll 1
    for (int nt = 0; nt < 16; ++nt) {
        float bias = b1[nt * 16 + l];
        v8f acc = {bias, bias, bias, bias, bias, bias, bias, bias};
        #pragma unroll
        for (int kt = 0; kt < 4; ++kt) {
            v16h a = load_a_frag(buf0, DIN, kt, hh, l);
            v16h b = load_b_frag(pw1, 16, kt, nt, lane);
            acc = __builtin_amdgcn_wmma_f32_16x16x32_f16(
                      false, a, false, b, (short)0, acc, false, false);
        }
        #pragma unroll
        for (int j = 0; j < 8; ++j) {
            // C/D layout: (row = j + 8*hh, col = nt*16 + l)
            buf1[(j + hh * 8) * DHID + nt * 16 + l] = (_Float16)gelu_exact(acc[j]);
        }
    }

    // -------- layer 2: h2 = gelu(h1 @ W2(256x256) + b2) -> buf0 -------------
    #pragma unroll 1
    for (int nt = 0; nt < 16; ++nt) {
        float bias = b2[nt * 16 + l];
        v8f acc = {bias, bias, bias, bias, bias, bias, bias, bias};
        #pragma unroll
        for (int kt = 0; kt < 8; ++kt) {
            v16h a = load_a_frag(buf1, DHID, kt, hh, l);
            v16h b = load_b_frag(pw2, 16, kt, nt, lane);
            acc = __builtin_amdgcn_wmma_f32_16x16x32_f16(
                      false, a, false, b, (short)0, acc, false, false);
        }
        #pragma unroll
        for (int j = 0; j < 8; ++j) {
            buf0[(j + hh * 8) * DHID + nt * 16 + l] = (_Float16)gelu_exact(acc[j]);
        }
    }

    // -------- layer 3: h3 = h2 @ W3(256x128) + b3 (kept in registers) -------
    v8f acc3[8];
    #pragma unroll 1
    for (int nt = 0; nt < 8; ++nt) {
        float bias = b3[nt * 16 + l];
        v8f acc = {bias, bias, bias, bias, bias, bias, bias, bias};
        #pragma unroll
        for (int kt = 0; kt < 8; ++kt) {
            v16h a = load_a_frag(buf0, DHID, kt, hh, l);
            v16h b = load_b_frag(pw3, 8, kt, nt, lane);
            acc = __builtin_amdgcn_wmma_f32_16x16x32_f16(
                      false, a, false, b, (short)0, acc, false, false);
        }
        acc3[nt] = acc;
    }

    // -------- LayerNorm over 128 features per row ---------------------------
    float lg[8], lb[8];
    #pragma unroll
    for (int nt = 0; nt < 8; ++nt) {
        lg[nt] = lng[nt * 16 + l];
        lb[nt] = lnb[nt * 16 + l];
    }
    #pragma unroll
    for (int j = 0; j < 8; ++j) {
        float s = 0.f, q = 0.f;
        #pragma unroll
        for (int nt = 0; nt < 8; ++nt) {
            float v = acc3[nt][j];
            s += v; q += v * v;
        }
        // butterfly over the 16 lanes of this half-wave (masks < 16)
        #pragma unroll
        for (int m = 1; m <= 8; m <<= 1) {
            s += __shfl_xor(s, m, 32);
            q += __shfl_xor(q, m, 32);
        }
        float mean = s * (1.0f / 128.0f);
        float var  = q * (1.0f / 128.0f) - mean * mean;
        float rstd = rsqrtf(var + 1e-5f);
        int gr = row0 + j + hh * 8;
        if (gr < NN) {
            #pragma unroll
            for (int nt = 0; nt < 8; ++nt) {
                out[(long long)gr * DOUT + nt * 16 + l] =
                    (acc3[nt][j] - mean) * rstd * lg[nt] + lb[nt];
            }
        }
    }
}

// ---------------------------------------------------------------- launcher
extern "C" void kernel_launch(void* const* d_in, const int* in_sizes, int n_in,
                              void* d_out, int out_size, void* d_ws, size_t ws_size,
                              hipStream_t stream) {
    const float* x   = (const float*)d_in[0];
    const int*   ei  = (const int*)  d_in[1];
    const float* W1  = (const float*)d_in[2];
    const float* b1  = (const float*)d_in[3];
    const float* W2  = (const float*)d_in[4];
    const float* b2  = (const float*)d_in[5];
    const float* W3  = (const float*)d_in[6];
    const float* b3  = (const float*)d_in[7];
    const float* lng = (const float*)d_in[8];
    const float* lnb = (const float*)d_in[9];
    float* out = (float*)d_out;

    char* ws = (char*)d_ws;
    float*    agg = (float*)ws;                                   // 25.6 MB
    _Float16* pw1 = (_Float16*)(ws + (size_t)NN * DIN * 4);       // 64 KB
    _Float16* pw2 = pw1 + DIN * DHID;                             // 128 KB
    _Float16* pw3 = pw2 + DHID * DHID;                            // 64 KB

    // 1) zero accumulator
    int n4 = NN * DIN / 4;
    zero_f4_kernel<<<(n4 + 255) / 256, 256, 0, stream>>>((float4*)agg, n4);

    // 2) pack weights into WMMA-B swizzled f16
    pack_weight_kernel<<<(DIN * DHID + 255) / 256, 256, 0, stream>>>(W1, pw1, DIN, DHID);
    pack_weight_kernel<<<(DHID * DHID + 255) / 256, 256, 0, stream>>>(W2, pw2, DHID, DHID);
    pack_weight_kernel<<<(DHID * DOUT + 255) / 256, 256, 0, stream>>>(W3, pw3, DHID, DOUT);

    // 3) bidirectional scatter-add (one 32-lane group per edge)
    long long threads = (long long)NEDGE * 32;
    scatter_kernel<<<(unsigned)((threads + 255) / 256), 256, 0, stream>>>(x, ei, agg, NEDGE);

    // 4) fused WMMA MLP + GELU + LayerNorm (64 rows per 128-thread block)
    mlp_kernel<<<(NN + 63) / 64, 128, 0, stream>>>(agg, pw1, pw2, pw3,
                                                   b1, b2, b3, lng, lnb, out);
}
